// GeneralizedMamba2_49598282334525
// MI455X (gfx1250) — matrix operand order repaired and checked by reading, not compiled, not run on hardware
//
#include <hip/hip_runtime.h>

// ---------------------------------------------------------------------------
// GeneralizedMamba2 on gfx1250 (MI455X).
// Phase 1/5: bf16 WMMA GEMMs with 4x4 register tiling (64x64 per wave).
// Phase 2:   depthwise causal conv.
// Phase 3:   per-(batch,head) recurrent scan; state in registers, R^T in LDS,
//            next-step inputs prefetched via GLOBAL_LOAD_ASYNC_TO_LDS_B32
//            (ASYNCcnt double-buffering).
// Phase 4:   gate fused with f32->bf16 convert.
// ---------------------------------------------------------------------------

#define D_MODEL   768
#define D_INNER   1536
#define HEADDIM   64
#define D_STATE   64
#define NHEADS    24
#define CONV_DIM  4608          // NHEADS * 192
#define CONV_PH   192           // HEADDIM + 2*D_STATE
#define D_PROJ    6168          // D_INNER + CONV_DIM + NHEADS
#define DT_COL    6144          // D_INNER + CONV_DIM

typedef __attribute__((ext_vector_type(16))) __bf16 v16bf;
typedef __attribute__((ext_vector_type(8)))  __bf16 v8bf;
typedef __attribute__((ext_vector_type(4)))  __bf16 v4bf;
typedef __attribute__((ext_vector_type(8)))  float  v8f;

__device__ __forceinline__ float siluf(float v) {
    return v * (1.0f / (1.0f + __expf(-v)));
}
__device__ __forceinline__ float softplusf(float v) {
    return (v > 20.0f) ? v : __logf(1.0f + __expf(v));
}

// ---------------------------------------------------------------------------
// f32 -> bf16 pre-pass (4 elements/thread, vector stores).
// ---------------------------------------------------------------------------
__global__ __launch_bounds__(256)
void f32_to_bf16_kernel(const float* __restrict__ in, __bf16* __restrict__ out, int n)
{
    int i = (blockIdx.x * 256 + threadIdx.x) * 4;
    if (i + 3 < n) {
        float4 v = *(const float4*)(in + i);
        v4bf o;
        o[0] = (__bf16)v.x; o[1] = (__bf16)v.y; o[2] = (__bf16)v.z; o[3] = (__bf16)v.w;
        *(v4bf*)(out + i) = o;
    } else {
        for (; i < n; ++i) out[i] = (__bf16)in[i];
    }
}

// ---------------------------------------------------------------------------
// C(MxN) f32 = A(MxK) @ B(NxK)^T with bf16 operands.
// Block = 128 threads = 4 waves, each wave computes a 64x64 tile (4x4 WMMA
// fragments); block tile = 128x128. M must be a multiple of 128. N tail is
// handled with clamped B loads + predicated stores (EXEC all-ones for WMMA).
// Inner loop per K-step: 16x global_load_b128 -> 16x v_wmma (no cvt).
// ---------------------------------------------------------------------------
__global__ __launch_bounds__(128)
void gemm_abT_wmma_bf16(const __bf16* __restrict__ A,
                        const __bf16* __restrict__ B,
                        float* __restrict__ C,
                        int M, int N, int K)
{
    const int lane = threadIdx.x & 31;
    const int wave = threadIdx.x >> 5;        // 0..3
    const int lo   = lane & 15;
    const int hi   = lane >> 4;               // 0 or 1
    const int mWave = blockIdx.y * 128 + (wave >> 1) * 64;
    const int nWave = blockIdx.x * 128 + (wave & 1) * 64;
    if (nWave >= N) return;                   // wave-uniform

    // A: lane holds row m, elems 0-7 <-> K=8*hi+0..7, elems 8-15 <-> +16..23
    const __bf16* Ap[4];
    #pragma unroll
    for (int i = 0; i < 4; ++i)
        Ap[i] = A + (size_t)(mWave + i * 16 + lo) * K + hi * 8;
    // B: lane holds column n, elems 0-7 <-> K=8*hi+0..7, elems 8-15 <-> +16..23
    const __bf16* Bp[4];
    #pragma unroll
    for (int j = 0; j < 4; ++j) {
        int n = nWave + j * 16 + lo;
        if (n > N - 1) n = N - 1;             // clamp OOB columns
        Bp[j] = B + (size_t)n * K + hi * 8;
    }

    v8f acc[4][4] = {};

    for (int k0 = 0; k0 < K; k0 += 32) {
        v16bf a[4], b[4];
        #pragma unroll
        for (int i = 0; i < 4; ++i) {
            v8bf l8 = *(const v8bf*)(Ap[i] + k0);
            v8bf h8 = *(const v8bf*)(Ap[i] + k0 + 16);
            a[i] = __builtin_shufflevector(l8, h8,
                       0,1,2,3,4,5,6,7,8,9,10,11,12,13,14,15);
        }
        #pragma unroll
        for (int j = 0; j < 4; ++j) {
            v8bf l8 = *(const v8bf*)(Bp[j] + k0);
            v8bf h8 = *(const v8bf*)(Bp[j] + k0 + 16);
            b[j] = __builtin_shufflevector(l8, h8,
                       0,1,2,3,4,5,6,7,8,9,10,11,12,13,14,15);
        }

        if (k0 + 64 < K) {                    // gfx1250 prefetch path
            __builtin_prefetch(Ap[0] + k0 + 64, 0, 1);
            __builtin_prefetch(Bp[0] + k0 + 64, 0, 1);
        }

        #pragma unroll
        for (int i = 0; i < 4; ++i)
            #pragma unroll
            for (int j = 0; j < 4; ++j)
                acc[i][j] = __builtin_amdgcn_wmma_f32_16x16x32_bf16(
                                false, a[i], false, b[j], (short)0,
                                acc[i][j], false, false);
    }

    // D layout: VGPR r -> M = r + 8*hi, N = lo
    #pragma unroll
    for (int i = 0; i < 4; ++i) {
        #pragma unroll
        for (int j = 0; j < 4; ++j) {
            int n = nWave + j * 16 + lo;
            if (n < N) {
                float* cp = C + (size_t)(mWave + i * 16 + hi * 8) * N + n;
                #pragma unroll
                for (int r = 0; r < 8; ++r)
                    cp[(size_t)r * N] = acc[i][j][r];
            }
        }
    }
}

// ---------------------------------------------------------------------------
// Depthwise causal conv (kernel=4, left pad 3) over the xBC slice of zxbcdt.
// ---------------------------------------------------------------------------
__global__ __launch_bounds__(256)
void conv1d_kernel(const float* __restrict__ zx,
                   const float* __restrict__ w,     // (4608, 1, 4)
                   const float* __restrict__ bias,  // (4608)
                   float* __restrict__ xc,
                   int L, int total)
{
    int idx = blockIdx.x * 256 + threadIdx.x;
    if (idx >= total) return;
    int c  = idx % CONV_DIM;
    int bl = idx / CONV_DIM;
    int l  = bl % L;

    const float* col = zx + (size_t)bl * D_PROJ + D_INNER + c;
    const float* wc  = w + c * 4;
    float acc = bias[c];
    #pragma unroll
    for (int j = 0; j < 4; ++j) {
        int d = j - 3;
        if (l + d >= 0)
            acc = __fmaf_rn(col[(ptrdiff_t)d * D_PROJ], wc[j], acc);
    }
    xc[idx] = acc;
}

// ---------------------------------------------------------------------------
// Async prefetch of one 192-float (768 B) timestep tile into LDS.
// Thread t covers dwords t, t+64, t+128: the same INST_OFFSET applies to both
// the LDS and global address, and both buffers are contiguous, so three
// global_load_async_to_lds_b32 with offsets 0/256/512 move the whole tile.
// Each thread later reads back exactly the dwords it issued, so a per-wave
// s_wait_asynccnt is sufficient (no barrier needed for this data).
// ---------------------------------------------------------------------------
__device__ __forceinline__ void async_tile_load(const float* gsrc, float* ldst, int t)
{
    unsigned int       l0 = (unsigned int)(uintptr_t)(ldst + t);
    unsigned long long g0 = (unsigned long long)(uintptr_t)(gsrc + t);
    asm volatile("global_load_async_to_lds_b32 %0, %1, off"
                 :: "v"(l0), "v"(g0) : "memory");
    asm volatile("global_load_async_to_lds_b32 %0, %1, off offset:256"
                 :: "v"(l0), "v"(g0) : "memory");
    asm volatile("global_load_async_to_lds_b32 %0, %1, off offset:512"
                 :: "v"(l0), "v"(g0) : "memory");
}

// ---------------------------------------------------------------------------
// Recurrent scan. One workgroup (64 threads = 2 wave32) per (b,h) chain.
// Thread t owns state row p=t (64 f32 in registers) and lane n=t of B/C.
// R matrices staged transposed in LDS with stride 65 (conflict-free).
// Next-step x/B/C tile double-buffered in LDS via async loads.
// ---------------------------------------------------------------------------
__global__ __launch_bounds__(64)
void scan_kernel(const float* __restrict__ xc,     // (B*L, 4608)
                 const float* __restrict__ zx,     // (B*L, 6168) (dt_raw col)
                 const float* __restrict__ Rx,     // (24,64,64)
                 const float* __restrict__ RB,
                 const float* __restrict__ RC,
                 const float* __restrict__ Rdt,    // (24,64)
                 const float* __restrict__ dt_bias,
                 const float* __restrict__ A_log,
                 const float* __restrict__ Dw,     // (24,64)
                 float* __restrict__ y,            // (B*L, 1536)
                 int L)
{
    const int b = blockIdx.x / NHEADS;
    const int h = blockIdx.x % NHEADS;
    const int t = threadIdx.x;

    __shared__ float RxT[64 * 65];
    __shared__ float RBT[64 * 65];
    __shared__ float RCT[64 * 65];
    __shared__ float Rd[64], ys[64], Bs[64], Cs[64];
    __shared__ float xtile[2][CONV_PH];            // double-buffered step tile

    const float* xbc0 = xc + (size_t)(b * L) * CONV_DIM + h * CONV_PH;

    // prime the pipeline: async-load step 0 while we stage R^T into LDS
    async_tile_load(xbc0, xtile[0], t);

    const float* rxh = Rx + (size_t)h * 4096;
    const float* rbh = RB + (size_t)h * 4096;
    const float* rch = RC + (size_t)h * 4096;
    for (int j = 0; j < 64; ++j) {        // T[q*65+p] = R[p*64+q]; p=j, q=t
        RxT[t * 65 + j] = rxh[j * 64 + t];
        RBT[t * 65 + j] = rbh[j * 64 + t];
        RCT[t * 65 + j] = rch[j * 64 + t];
    }
    Rd[t] = Rdt[h * 64 + t];
    ys[t] = 0.0f;

    float st[64];
    #pragma unroll
    for (int n = 0; n < 64; ++n) st[n] = 0.0f;

    const float dtb  = dt_bias[h];
    const float Aneg = -__expf(A_log[h]);
    const float Dh   = Dw[h * 64 + t];
    __syncthreads();

    for (int l = 0; l < L; ++l) {
        const size_t row = (size_t)(b * L + l);
        // kick off next step's tile, then retire this step's 3 async loads
        if (l + 1 < L) {
            async_tile_load(xbc0 + (size_t)(l + 1) * CONV_DIM,
                            xtile[(l + 1) & 1], t);
            asm volatile("s_wait_asynccnt 0x3" ::: "memory");
        } else {
            asm volatile("s_wait_asynccnt 0x0" ::: "memory");
        }

        const float* xrow = xtile[l & 1];
        float x_u = xrow[t];
        float B_u = xrow[64 + t];
        float C_u = xrow[128 + t];
        float dtr = zx[row * D_PROJ + DT_COL + h];     // scalar broadcast

        float rx = 0.f, rb = 0.f, rc = 0.f, rd = 0.f;
        #pragma unroll 8
        for (int q = 0; q < 64; ++q) {
            float yq = ys[q];                           // LDS broadcast
            rx = __fmaf_rn(RxT[q * 65 + t], yq, rx);
            rb = __fmaf_rn(RBT[q * 65 + t], yq, rb);
            rc = __fmaf_rn(RCT[q * 65 + t], yq, rc);
            rd = __fmaf_rn(Rd[q], yq, rd);              // redundant scalar dot
        }
        float xv = siluf(x_u + rx);
        float Bm = siluf(B_u + rb);
        float Cm = siluf(C_u + rc);
        float dt = softplusf(dtr + rd + dtb);
        float dA = __expf(dt * Aneg);

        __syncthreads();                  // all readers of ys/Bs/Cs done
        Bs[t] = Bm;
        Cs[t] = Cm;
        __syncthreads();                  // Bm/Cm visible

        float dtx  = dt * xv;
        float yacc = Dh * xv;
        #pragma unroll
        for (int n = 0; n < 64; ++n) {
            float s = __fmaf_rn(st[n], dA, dtx * Bs[n]);
            st[n] = s;
            yacc = __fmaf_rn(s, Cs[n], yacc);
        }
        y[row * D_INNER + h * 64 + t] = yacc;
        ys[t] = yacc;
        __syncthreads();                  // new ys visible for next step
    }
}

// ---------------------------------------------------------------------------
// Gate + convert: out_bf16 = bf16( y * silu(z) ) ; z = zxbcdt[..., :D_INNER]
// ---------------------------------------------------------------------------
__global__ __launch_bounds__(256)
void gate_cvt_kernel(const float* __restrict__ yb,
                     const float* __restrict__ zx,
                     __bf16* __restrict__ out, int total)
{
    int i = blockIdx.x * 256 + threadIdx.x;
    if (i >= total) return;
    int col = i % D_INNER;
    size_t row = (size_t)(i / D_INNER);
    float z = zx[row * D_PROJ + col];
    out[i] = (__bf16)(yb[i] * z * (1.0f / (1.0f + __expf(-z))));
}

// ---------------------------------------------------------------------------
extern "C" void kernel_launch(void* const* d_in, const int* in_sizes, int n_in,
                              void* d_out, int out_size, void* d_ws, size_t ws_size,
                              hipStream_t stream)
{
    const float* u       = (const float*)d_in[0];
    const float* W_in    = (const float*)d_in[1];
    const float* conv_w  = (const float*)d_in[2];
    const float* conv_b  = (const float*)d_in[3];
    const float* R_x     = (const float*)d_in[4];
    const float* R_B     = (const float*)d_in[5];
    const float* R_C     = (const float*)d_in[6];
    const float* R_dt    = (const float*)d_in[7];
    const float* dt_bias = (const float*)d_in[8];
    const float* A_log   = (const float*)d_in[9];
    const float* Dw      = (const float*)d_in[10];
    const float* W_out   = (const float*)d_in[11];
    float* out = (float*)d_out;

    const int BL = in_sizes[0] / D_MODEL;   // B*L = 4096
    const int B  = 2;
    const int L  = BL / B;

    // workspace carve-up (all slice sizes are multiples of 512 bytes)
    char* ws = (char*)d_ws;
    float*  zx      = (float*)ws;                 ws += (size_t)BL * D_PROJ   * 4;
    float*  xcv     = (float*)ws;                 ws += (size_t)BL * CONV_DIM * 4;
    float*  yb      = (float*)ws;                 ws += (size_t)BL * D_INNER  * 4;
    __bf16* u_bf    = (__bf16*)ws;                ws += (size_t)BL * D_MODEL  * 2;
    __bf16* Win_bf  = (__bf16*)ws;                ws += (size_t)D_PROJ * D_MODEL * 2;
    __bf16* ybg_bf  = (__bf16*)ws;                ws += (size_t)BL * D_INNER  * 2;
    __bf16* Wout_bf = (__bf16*)ws;

    // 0) convert GEMM operands to bf16 once
    {
        int n = BL * D_MODEL;
        f32_to_bf16_kernel<<<(n / 4 + 255) / 256, 256, 0, stream>>>(u, u_bf, n);
        n = D_PROJ * D_MODEL;
        f32_to_bf16_kernel<<<(n / 4 + 255) / 256, 256, 0, stream>>>(W_in, Win_bf, n);
        n = D_MODEL * D_INNER;
        f32_to_bf16_kernel<<<(n / 4 + 255) / 256, 256, 0, stream>>>(W_out, Wout_bf, n);
    }
    // 1) zxbcdt = u @ W_in^T   (M=BL, N=6168, K=768)
    {
        dim3 grid((D_PROJ + 127) / 128, BL / 128);
        gemm_abT_wmma_bf16<<<grid, 128, 0, stream>>>(u_bf, Win_bf, zx, BL, D_PROJ, D_MODEL);
    }
    // 2) depthwise causal conv on xBC slice
    {
        int total = BL * CONV_DIM;
        conv1d_kernel<<<(total + 255) / 256, 256, 0, stream>>>(zx, conv_w, conv_b, xcv, L, total);
    }
    // 3) recurrent scan, one workgroup per (b,h)
    scan_kernel<<<B * NHEADS, 64, 0, stream>>>(xcv, zx, R_x, R_B, R_C, R_dt,
                                               dt_bias, A_log, Dw, yb, L);
    // 4) gate + bf16 convert
    {
        int total = BL * D_INNER;
        gate_cvt_kernel<<<(total + 255) / 256, 256, 0, stream>>>(yb, zx, ybg_bf, total);
    }
    // 5) out = y_gated @ W_out^T   (M=BL, N=768, K=1536)
    {
        dim3 grid(D_MODEL / 128, BL / 128);
        gemm_abT_wmma_bf16<<<grid, 128, 0, stream>>>(ybg_bf, Wout_bf, out, BL, D_MODEL, D_INNER);
    }
}
